// FolkLSTM_42193758715963
// MI455X (gfx1250) — compile-verified, matrix operand order checked by reading, not run on hardware
//
#include <hip/hip_runtime.h>

// ---------------- types ----------------
typedef __attribute__((ext_vector_type(16))) __bf16   v16bf;
typedef __attribute__((ext_vector_type(8)))  float    v8f;
typedef __attribute__((ext_vector_type(4)))  unsigned v4u;
typedef __attribute__((ext_vector_type(4)))  int      v4i;

struct Frag { v4u q[2]; };                       // 32 bytes = 16 bf16
__device__ __forceinline__ v16bf asbf(const Frag& f) {
    return __builtin_bit_cast(v16bf, f);
}

__device__ __forceinline__ unsigned short f2bf(float f) {
    unsigned u = __float_as_uint(f);
    u = (u + 0x7FFFu + ((u >> 16) & 1u)) >> 16;  // round-to-nearest-even
    return (unsigned short)u;
}
__device__ __forceinline__ float sigm(float x) {
    return 1.f / (1.f + __expf(-x));
}
__device__ __forceinline__ float tanhr(float x) {
    x = fminf(15.f, fmaxf(-15.f, x));
    float e = __expf(2.f * x);
    return (e - 1.f) / (e + 1.f);
}

#define BT    1024
#define BB    64
#define UU    512
#define VV    512
#define EE    256
#define LDSLD 1032      // 1024 + 8 bf16 pad -> 2064B row stride, kills bank conflicts

// ---------------- CDNA5 async global->LDS copy (ASYNCcnt path) ----------------
#if defined(__has_builtin)
#  if __has_builtin(__builtin_amdgcn_global_load_async_to_lds_b128)
#    define HAVE_ASYNC_B128 1
#  endif
#  if __has_builtin(__builtin_amdgcn_s_wait_asynccnt)
#    define HAVE_WAIT_ASYNC 1
#  endif
#endif

__device__ __forceinline__ void async_cp16(unsigned short* lds_dst, const unsigned short* gsrc) {
#ifdef HAVE_ASYNC_B128
    __builtin_amdgcn_global_load_async_to_lds_b128(
        (__attribute__((address_space(1))) v4i*)gsrc,
        (__attribute__((address_space(3))) v4i*)lds_dst, 0, 0);
#else
    *(v4u*)lds_dst = *(const v4u*)gsrc;
#endif
}
__device__ __forceinline__ void async_wait0() {
#ifdef HAVE_ASYNC_B128
#  ifdef HAVE_WAIT_ASYNC
    __builtin_amdgcn_s_wait_asynccnt(0);
#  else
    asm volatile("s_wait_asynccnt 0x0" ::: "memory");
#  endif
#endif
}

// ---------------- grid barrier (monotonic, poison-tolerant) ----------------
__device__ __forceinline__ void grid_barrier(unsigned* bar, int nblk) {
    __syncthreads();
    if (threadIdx.x == 0) {
        __threadfence();
        unsigned* arrive  = bar;
        unsigned* release = bar + 1;
        unsigned snap = __hip_atomic_load(release, __ATOMIC_ACQUIRE, __HIP_MEMORY_SCOPE_AGENT);
        unsigned a = __hip_atomic_fetch_add(arrive, 1u, __ATOMIC_ACQ_REL, __HIP_MEMORY_SCOPE_AGENT);
        if (((a + 1u) % (unsigned)nblk) == 0u)
            __hip_atomic_fetch_add(release, 1u, __ATOMIC_ACQ_REL, __HIP_MEMORY_SCOPE_AGENT);
        while (__hip_atomic_load(release, __ATOMIC_ACQUIRE, __HIP_MEMORY_SCOPE_AGENT) == snap)
            __builtin_amdgcn_s_sleep(2);
        __threadfence();
    }
    __syncthreads();
}

// ---------------- conversion kernels ----------------
__global__ void conv_bf16_kernel(const float* __restrict__ src,
                                 unsigned short* __restrict__ dst, int n) {
    int i = blockIdx.x * 256 + threadIdx.x;
    if (i < n) dst[i] = f2bf(src[i]);
}

// dst[n][k] = (k < Kx ? Wk[k][n] : Wr[k-Kx][n])  transposed combined weight, bf16
__global__ void conv_wt_kernel(const float* __restrict__ Wk, const float* __restrict__ Wr,
                               unsigned short* __restrict__ dst, int Kx, int Kl) {
    int i = blockIdx.x * 256 + threadIdx.x;
    if (i >= 2048 * Kl) return;
    int n = i / Kl, k = i % Kl;
    float v = (k < Kx) ? Wk[k * 2048 + n] : Wr[(k - Kx) * 2048 + n];
    dst[i] = f2bf(v);
}

__global__ void conv_wdt_kernel(const float* __restrict__ Wd, unsigned short* __restrict__ dst) {
    int i = blockIdx.x * 256 + threadIdx.x;   // 512*512
    int n = i >> 9, k = i & 511;
    dst[i] = f2bf(Wd[k * VV + n]);
}

// ---------------- persistent 3-layer LSTM kernel ----------------
// grid = 32 WGs x 256 threads (8 waves). WG wg owns hidden units [wg*16, wg*16+16).
__global__ __launch_bounds__(256, 1) void lstm_kernel(
    const int* __restrict__ tokens, const int* __restrict__ lengths,
    const unsigned short* __restrict__ emb_bf,
    const unsigned short* __restrict__ W0t, const unsigned short* __restrict__ W1t,
    const unsigned short* __restrict__ W2t,
    const float* __restrict__ b0, const float* __restrict__ b1, const float* __restrict__ b2,
    unsigned short* __restrict__ XA, unsigned short* __restrict__ XB,
    unsigned short* __restrict__ h_glob, unsigned* __restrict__ bar)
{
    __shared__ unsigned short xh[BB][LDSLD];   // [x_t | h] staged, bf16 (132KB)
    __shared__ unsigned short wl[BB][LDSLD];   // layer-resident weight slice (132KB)
    __shared__ float zl[4][BB][16];            // gate pre-activations (16KB)
    __shared__ float c_l[BB][16];              // persistent cell-state slice (4KB)
    __shared__ float h_l[BB][16];              // fp32 copy of own h slice (4KB)

    const int tid  = threadIdx.x;
    const int lane = tid & 31;
    const int wave = tid >> 5;
    const int j0   = blockIdx.x * 16;
    const int nblk = gridDim.x;

    for (int l = 0; l < 3; ++l) {
        const int Kx = (l == 0) ? EE : UU;
        const int Kl = Kx + UU;
        const unsigned short* Wt  = (l == 0) ? W0t : (l == 1) ? W1t : W2t;
        const float* bias         = (l == 0) ? b0  : (l == 1) ? b1  : b2;
        const unsigned short* Xin = (l == 1) ? XB : XA;   // unused for l==0
        unsigned short* Xout      = (l == 1) ? XA : XB;   // l0->XB, l1->XA, l2->XB

        // ---- layer init: preload weight slice into LDS (reused for 1024 steps) ----
        // local row lr = gate*16 + nl  <->  global row n = gate*512 + j0 + nl
        {
            const int cpr = Kl >> 3;                    // 16B chunks per row
            for (int i = tid; i < 64 * cpr; i += 256) {
                int lr = i / cpr, q = i - lr * cpr;
                int n  = ((lr >> 4) * UU) + j0 + (lr & 15);
                async_cp16(&wl[lr][q * 8], Wt + (size_t)n * Kl + q * 8);
            }
            async_wait0();
        }
        // init own h/c slice + own columns of global h
        for (int e = tid; e < BB * 16; e += 256) {
            int m = e >> 4, n = e & 15;
            c_l[m][n] = 0.f;  h_l[m][n] = 0.f;
            h_glob[m * UU + j0 + n] = 0;
        }
        grid_barrier(bar, nblk);

        for (int t = 0; t < BT; ++t) {
            // ---- phase 0: stage [x_t | h] into LDS via async copies ----
            if (l == 0) {
                for (int i = tid; i < 64 * 32; i += 256) {        // 64 rows x 32 chunks
                    int b = i >> 5, q = i & 31;
                    int tk = tokens[b * BT + t];
                    async_cp16(&xh[b][q * 8], emb_bf + (size_t)tk * EE + q * 8);
                }
            } else {
                const unsigned short* src = Xin + (size_t)t * BB * UU;
                for (int i = tid; i < 64 * 64; i += 256) {
                    int b = i >> 6, q = i & 63;
                    async_cp16(&xh[b][q * 8], src + b * UU + q * 8);
                }
            }
            for (int i = tid; i < 64 * 64; i += 256) {            // h part
                int b = i >> 6, q = i & 63;
                async_cp16(&xh[b][Kx + q * 8], h_glob + b * UU + q * 8);
            }
            async_wait0();
            __syncthreads();

            // ---- phase 1: WMMA  z = [x|h] @ [Wk;Wr]^slice  (all operands in LDS) ----
            const int mt = wave >> 1;              // M tile 0..3 (batch rows)
            const int g0 = (wave & 1) * 2;         // gate pair {0,1} or {2,3}
            const int ar = mt * 16 + (lane & 15);
            const int kh = (lane >> 4) ? 8 : 0;
            const unsigned short* bp0 = &wl[(g0    ) * 16 + (lane & 15)][0];
            const unsigned short* bp1 = &wl[(g0 + 1) * 16 + (lane & 15)][0];
            v8f acc0 = {}; v8f acc1 = {};
            for (int kb = 0; kb < Kl; kb += 32) {
                int k = kb + kh;
                Frag a, bf0, bf1;
                a.q[0]   = *(const v4u*)&xh[ar][k];
                a.q[1]   = *(const v4u*)&xh[ar][k + 16];
                bf0.q[0] = *(const v4u*)&bp0[k];
                bf0.q[1] = *(const v4u*)&bp0[k + 16];
                bf1.q[0] = *(const v4u*)&bp1[k];
                bf1.q[1] = *(const v4u*)&bp1[k + 16];
                acc0 = __builtin_amdgcn_wmma_f32_16x16x32_bf16(
                           false, asbf(a), false, asbf(bf0), (short)0, acc0, false, false);
                acc1 = __builtin_amdgcn_wmma_f32_16x16x32_bf16(
                           false, asbf(a), false, asbf(bf1), (short)0, acc1, false, false);
            }
            // ---- phase 2: spill z tiles to LDS ----
            #pragma unroll
            for (int v = 0; v < 8; ++v) {
                int m = v + ((lane >> 4) << 3);
                zl[g0    ][mt * 16 + m][lane & 15] = acc0[v];
                zl[g0 + 1][mt * 16 + m][lane & 15] = acc1[v];
            }
            __syncthreads();

            // ---- phase 3: fused gates + state update + masking ----
            {
                int e0 = tid * 4;
                #pragma unroll
                for (int e = e0; e < e0 + 4; ++e) {
                    int m = e >> 4, n = e & 15;
                    int j = j0 + n;
                    float zi = zl[0][m][n] + bias[j];
                    float zf = zl[1][m][n] + bias[UU + j];
                    float zg = zl[2][m][n] + bias[2 * UU + j];
                    float zo = zl[3][m][n] + bias[3 * UU + j];
                    float i_ = sigm(zi), f_ = sigm(zf), g_ = tanhr(zg), o_ = sigm(zo);
                    float c  = c_l[m][n];
                    float cn = f_ * c + i_ * g_;
                    float hn = o_ * tanhr(cn);
                    bool  mk = t < lengths[m];
                    float c2 = mk ? cn : c;
                    float h2 = mk ? hn : h_l[m][n];
                    float ot = mk ? hn : 0.f;
                    c_l[m][n] = c2;  h_l[m][n] = h2;
                    h_glob[m * UU + j] = f2bf(h2);
                    Xout[((size_t)t * BB + m) * UU + j] = f2bf(ot);
                }
            }
            grid_barrier(bar, nblk);
        }
    }
}

// ---------------- decoder: logits[b][t][:] = X[t][b][:] @ Wd + bd ----------------
__global__ __launch_bounds__(256) void decoder_kernel(
    const unsigned short* __restrict__ X, const unsigned short* __restrict__ Wdt,
    const float* __restrict__ bd, float* __restrict__ out)
{
    int tid = threadIdx.x, lane = tid & 31, wave = tid >> 5;
    int tile = blockIdx.x * 8 + wave;          // 131072 tiles = 4096 x 32
    int rt = tile >> 5, nt = tile & 31;
    int r  = rt * 16 + (lane & 15);
    int n  = nt * 16 + (lane & 15);
    int kh = (lane >> 4) ? 8 : 0;
    const unsigned short* ap = X   + (size_t)r * UU;
    const unsigned short* bp = Wdt + (size_t)n * UU;
    v8f acc = {};
    for (int kb = 0; kb < UU; kb += 32) {
        int k = kb + kh;
        Frag a, b;
        a.q[0] = *(const v4u*)&ap[k];  a.q[1] = *(const v4u*)&ap[k + 16];
        b.q[0] = *(const v4u*)&bp[k];  b.q[1] = *(const v4u*)&bp[k + 16];
        acc = __builtin_amdgcn_wmma_f32_16x16x32_bf16(
                  false, asbf(a), false, asbf(b), (short)0, acc, false, false);
    }
    float bias = bd[n];
    #pragma unroll
    for (int v = 0; v < 8; ++v) {
        int m  = v + ((lane >> 4) << 3);
        int r2 = rt * 16 + m;
        int t  = r2 >> 6, b = r2 & 63;
        out[(((size_t)b << 10) + t) * (size_t)VV + n] = acc[v] + bias;
    }
}

// ---------------- host launcher ----------------
extern "C" void kernel_launch(void* const* d_in, const int* in_sizes, int n_in,
                              void* d_out, int out_size, void* d_ws, size_t ws_size,
                              hipStream_t stream) {
    (void)in_sizes; (void)n_in; (void)out_size; (void)ws_size;
    const int*   tokens = (const int*)d_in[0];
    const int*   lengths= (const int*)d_in[1];
    const float* emb = (const float*)d_in[2];
    const float* Wk0 = (const float*)d_in[3];
    const float* Wr0 = (const float*)d_in[4];
    const float* b0  = (const float*)d_in[5];
    const float* Wk1 = (const float*)d_in[6];
    const float* Wr1 = (const float*)d_in[7];
    const float* b1  = (const float*)d_in[8];
    const float* Wk2 = (const float*)d_in[9];
    const float* Wr2 = (const float*)d_in[10];
    const float* b2  = (const float*)d_in[11];
    const float* Wd  = (const float*)d_in[12];
    const float* bd  = (const float*)d_in[13];
    float* out = (float*)d_out;

    char* ws = (char*)d_ws;
    size_t off = 0;
    auto take = [&](size_t bytes) { char* p = ws + off; off = (off + bytes + 255) & ~(size_t)255; return p; };
    unsigned short* XA     = (unsigned short*)take((size_t)BT * BB * UU * 2);   // 64 MB
    unsigned short* XB     = (unsigned short*)take((size_t)BT * BB * UU * 2);   // 64 MB
    unsigned short* emb_bf = (unsigned short*)take((size_t)VV * EE * 2);
    unsigned short* W0t    = (unsigned short*)take((size_t)2048 * 768 * 2);
    unsigned short* W1t    = (unsigned short*)take((size_t)2048 * 1024 * 2);
    unsigned short* W2t    = (unsigned short*)take((size_t)2048 * 1024 * 2);
    unsigned short* Wdt    = (unsigned short*)take((size_t)VV * UU * 2);
    unsigned short* h_glob = (unsigned short*)take((size_t)BB * UU * 2);
    unsigned*       bar    = (unsigned*)take(256);

    // weight / embedding conversion (runs every call; deterministic, inputs untouched)
    conv_bf16_kernel<<<(VV * EE + 255) / 256, 256, 0, stream>>>(emb, emb_bf, VV * EE);
    conv_wt_kernel<<<(2048 * 768  + 255) / 256, 256, 0, stream>>>(Wk0, Wr0, W0t, 256, 768);
    conv_wt_kernel<<<(2048 * 1024 + 255) / 256, 256, 0, stream>>>(Wk1, Wr1, W1t, 512, 1024);
    conv_wt_kernel<<<(2048 * 1024 + 255) / 256, 256, 0, stream>>>(Wk2, Wr2, W2t, 512, 1024);
    conv_wdt_kernel<<<(VV * UU + 255) / 256, 256, 0, stream>>>(Wd, Wdt);

    // persistent recurrent kernel: all 3 layers, grid-sync per timestep
    lstm_kernel<<<32, 256, 0, stream>>>(tokens, lengths, emb_bf, W0t, W1t, W2t,
                                        b0, b1, b2, XA, XB, h_glob, bar);

    // final projection (reads XB = layer-2 output)
    decoder_kernel<<<(BT * BB / 16) * 32 / 8, 256, 0, stream>>>(XB, Wdt, bd, out);
}
